// graph_constructor_28278064677430
// MI455X (gfx1250) — matrix operand order repaired and checked by reading, not compiled, not run on hardware
//
#include <hip/hip_runtime.h>
#include <hip/hip_bf16.h>

#define NN    2048
#define FT    512
#define BSZ   8
#define DIMN  64
#define NLAY  3
#define TOPK  20
#define AS    (FT + 8)          // padded LDS row stride (ushorts) -> conflict-free ds_load_b128

typedef __attribute__((ext_vector_type(16))) __bf16 v16bf;
typedef __attribute__((ext_vector_type(8)))  float  v8f;

struct __align__(16) U128 { unsigned int w0, w1, w2, w3; };
struct __align__(16) F4   { float a, b, c, d; };

union FragBF { v16bf v; U128 q[2]; };

// ---------------------------------------------------------------- kernel 0
// Split f32 x into bf16 hi + bf16 lo (round-to-nearest-even) for bf16x3 WMMA.
__device__ __forceinline__ unsigned short f32_to_bf16_rne(float f) {
  unsigned u = __float_as_uint(f);
  unsigned r = u + 0x7FFFu + ((u >> 16) & 1u);
  return (unsigned short)(r >> 16);
}

__global__ void split_bf16_kernel(const float* __restrict__ x,
                                  unsigned short* __restrict__ hi,
                                  unsigned short* __restrict__ lo) {
  size_t i = (size_t)blockIdx.x * blockDim.x + threadIdx.x;
  float v = x[i];
  unsigned short h = f32_to_bf16_rne(v);
  float hf = __uint_as_float((unsigned)h << 16);
  hi[i] = h;
  lo[i] = f32_to_bf16_rne(v - hf);
}

// ---------------------------------------------------------------- kernel 1
// Per-node tanh(Linear) chain: n = tanh(n @ W[l]^T + b[l]), 3 layers, 2 chains.
__global__ void node_mlp_kernel(const float* __restrict__ emb1,
                                const float* __restrict__ emb2,
                                const long long* __restrict__ idx,
                                const float* __restrict__ w1, const float* __restrict__ b1,
                                const float* __restrict__ w2, const float* __restrict__ b2,
                                float* __restrict__ o1, float* __restrict__ o2) {
  __shared__ float s1[DIMN];
  __shared__ float s2[DIMN];
  const int j = threadIdx.x;
  const int node = blockIdx.x;
  const long long g = idx[node];
  s1[j] = emb1[(size_t)g * DIMN + j];
  s2[j] = emb2[(size_t)g * DIMN + j];
  __syncthreads();
  for (int l = 0; l < NLAY; ++l) {
    float a1 = b1[l * DIMN + j];
    float a2 = b2[l * DIMN + j];
    const float* W1 = w1 + (size_t)(l * DIMN + j) * DIMN;
    const float* W2 = w2 + (size_t)(l * DIMN + j) * DIMN;
    for (int k = 0; k < DIMN; ++k) {
      a1 += s1[k] * W1[k];
      a2 += s2[k] * W2[k];
    }
    a1 = tanhf(a1);
    a2 = tanhf(a2);
    __syncthreads();
    s1[j] = a1;
    s2[j] = a2;
    __syncthreads();
  }
  o1[(size_t)node * DIMN + j] = s1[j];
  o2[(size_t)node * DIMN + j] = s2[j];
}

// ---------------------------------------------------------------- kernel 2
// adj_cs = relu(tanh(n1 n2^T - n2 n1^T)), 2048x2048 f32 into workspace.
__global__ void adj_cs_kernel(const float* __restrict__ n1,
                              const float* __restrict__ n2,
                              float* __restrict__ adjcs) {
  __shared__ float a1[16][DIMN], a2[16][DIMN], c1[16][DIMN], c2[16][DIMN];
  const int tx = threadIdx.x, ty = threadIdx.y;
  const int tid = ty * 16 + tx;
  const int i0 = blockIdx.y * 16, j0 = blockIdx.x * 16;
  for (int t = tid; t < 16 * DIMN; t += 256) {
    int r = t >> 6, k = t & 63;
    a1[r][k] = n1[(size_t)(i0 + r) * DIMN + k];
    a2[r][k] = n2[(size_t)(i0 + r) * DIMN + k];
    c1[r][k] = n1[(size_t)(j0 + r) * DIMN + k];
    c2[r][k] = n2[(size_t)(j0 + r) * DIMN + k];
  }
  __syncthreads();
  float d1 = 0.f, d2 = 0.f;
  for (int k = 0; k < DIMN; ++k) {
    d1 += a1[ty][k] * c2[tx][k];
    d2 += a2[ty][k] * c1[tx][k];
  }
  float a = tanhf(d1 - d2);
  adjcs[(size_t)(i0 + ty) * NN + (j0 + tx)] = a > 0.f ? a : 0.f;
}

// ---------------------------------------------------------------- kernel 3
// Fused: per (batch, 16-row tile): bf16x3 WMMA x@x^T -> leaky(-INF diag) ->
// softmax -> tanh(+eye) -> +adj_cs -> top-20 mask -> masked store. All
// intermediates stay in LDS; adj0 never touches HBM except its 20/row survivors.
__launch_bounds__(256, 1)
__global__ void fused_adj_kernel(const unsigned short* __restrict__ xhi,
                                 const unsigned short* __restrict__ xlo,
                                 const float* __restrict__ adjcs,
                                 float* __restrict__ out) {
  __shared__ unsigned short sh_ah[16 * AS];   // 16.25 KB  A-tile hi
  __shared__ unsigned short sh_al[16 * AS];   // 16.25 KB  A-tile lo
  __shared__ float sh_s[16 * NN];             // 128 KB    score strip
  __shared__ int   sh_ti[16 * TOPK];
  __shared__ float sh_tv[16 * TOPK];

  const int tid  = threadIdx.x;
  const int lane = tid & 31;
  const int wave = tid >> 5;
  const int l16  = lane & 15;
  const int kh   = lane >> 4;
  const int b    = blockIdx.y;
  const int r0   = blockIdx.x * 16;

  // ---- load A row tile (hi/lo) into padded LDS
  {
    const size_t abase = ((size_t)b * NN + r0) * FT;
    for (int t = tid; t < 16 * (FT / 8); t += 256) {
      int m = t >> 6, ch = (t & 63) * 8;
      *(U128*)(sh_ah + m * AS + ch) = *(const U128*)(xhi + abase + (size_t)m * FT + ch);
      *(U128*)(sh_al + m * AS + ch) = *(const U128*)(xlo + abase + (size_t)m * FT + ch);
    }
  }
  __syncthreads();

  // ---- GEMM over 128 column tiles; 8 waves round-robin the tiles
  for (int ct = wave; ct < NN / 16; ct += 8) {
    const int c0 = ct * 16;
    v8f acc;
#pragma unroll
    for (int j = 0; j < 8; ++j) acc[j] = 0.0f;
    // B operand: this lane owns column node c0+l16; rows are contiguous in memory
    const unsigned short* bh = xhi + ((size_t)b * NN + c0 + l16) * FT;
    const unsigned short* bl = xlo + ((size_t)b * NN + c0 + l16) * FT;
#pragma unroll 4
    for (int kk = 0; kk < FT; kk += 32) {
      FragBF ah, al, bhF, blF;
      // A frag (16x32 bf16): lane row l16; e0..7 -> k=kk+8*kh.., e8..15 -> k=kk+16+8*kh..
      ah.q[0] = *(const U128*)(sh_ah + l16 * AS + kk + 8 * kh);
      ah.q[1] = *(const U128*)(sh_ah + l16 * AS + kk + 16 + 8 * kh);
      al.q[0] = *(const U128*)(sh_al + l16 * AS + kk + 8 * kh);
      al.q[1] = *(const U128*)(sh_al + l16 * AS + kk + 16 + 8 * kh);
      // B frag (32x16 bf16): e0..7 -> k=kk+8*kh.., e8..15 -> k=kk+16+8*kh.. for this lane's column
      const U128* pbh = (const U128*)(bh + kk + 8 * kh);
      const U128* pbl = (const U128*)(bl + kk + 8 * kh);
      bhF.q[0] = pbh[0]; bhF.q[1] = pbh[2];
      blF.q[0] = pbl[0]; blF.q[1] = pbl[2];
      // fp32-accurate product via bf16 split: hi*hi + hi*lo + lo*hi
      acc = __builtin_amdgcn_wmma_f32_16x16x32_bf16(false, ah.v, false, bhF.v, (short)0, acc, false, false);
      acc = __builtin_amdgcn_wmma_f32_16x16x32_bf16(false, ah.v, false, blF.v, (short)0, acc, false, false);
      acc = __builtin_amdgcn_wmma_f32_16x16x32_bf16(false, al.v, false, bhF.v, (short)0, acc, false, false);
    }
    // C/D layout: element j -> (m = j + 8*kh, n = l16). Apply -INF diag + leaky.
#pragma unroll
    for (int j = 0; j < 8; ++j) {
      int m = j + 8 * kh;
      int i = r0 + m;
      int c = c0 + l16;
      float v = acc[j];
      if (i == c) v -= 1.0e8f;
      v = v > 0.0f ? v : 0.01f * v;
      sh_s[m * NN + c] = v;
    }
  }
  __syncthreads();

  // ---- per row: softmax -> tanh(+eye) -> + adj_cs -> iterated top-20
  for (int rr = 0; rr < 2; ++rr) {
    const int m = wave * 2 + rr;
    const int i = r0 + m;
    float* row = sh_s + m * NN;

    float mx = -3.4e38f;
    for (int c = lane; c < NN; c += 32) mx = fmaxf(mx, row[c]);
#pragma unroll
    for (int off = 16; off > 0; off >>= 1) mx = fmaxf(mx, __shfl_xor(mx, off, 32));

    float sum = 0.0f;
    for (int c = lane; c < NN; c += 32) sum += __expf(row[c] - mx);
#pragma unroll
    for (int off = 16; off > 0; off >>= 1) sum += __shfl_xor(sum, off, 32);
    const float inv = 1.0f / sum;

    const float* acrow = adjcs + (size_t)i * NN;
    for (int c = lane; c < NN; c += 32) {
      float p = __expf(row[c] - mx) * inv;
      float a = tanhf(p + ((c == i) ? 1.0f : 0.0f));
      row[c] = a + acrow[c];
    }

    volatile float* vrow = row;  // same-wave LDS RAW kept in order
    for (int t = 0; t < TOPK; ++t) {
      float bv = -3.4e38f;
      int bi = 0;
      for (int c = lane; c < NN; c += 32) {
        float v = vrow[c];
        if (v > bv) { bv = v; bi = c; }
      }
#pragma unroll
      for (int off = 16; off > 0; off >>= 1) {
        float ov = __shfl_xor(bv, off, 32);
        int   oi = __shfl_xor(bi, off, 32);
        if (ov > bv || (ov == bv && oi < bi)) { bv = ov; bi = oi; }
      }
      if (lane == 0) { sh_ti[m * TOPK + t] = bi; sh_tv[m * TOPK + t] = bv; }
      if (lane == (bi & 31)) vrow[bi] = -3.4e38f;  // knock out; owner lane re-reads it
    }
  }
  __syncthreads();

  // ---- masked dense store: each element written exactly once (no race)
  float* outb = out + ((size_t)b * NN + r0) * NN;
  for (int m = 0; m < 16; ++m) {
    int   ti[TOPK];
    float tv[TOPK];
#pragma unroll
    for (int t = 0; t < TOPK; ++t) { ti[t] = sh_ti[m * TOPK + t]; tv[t] = sh_tv[m * TOPK + t]; }
    float* orow = outb + (size_t)m * NN;
    for (int c4 = tid * 4; c4 < NN; c4 += 256 * 4) {
      float o0 = 0.f, o1 = 0.f, o2 = 0.f, o3 = 0.f;
#pragma unroll
      for (int t = 0; t < TOPK; ++t) {
        int ix = ti[t];
        float v = tv[t];
        o0 = (ix == c4    ) ? v : o0;
        o1 = (ix == c4 + 1) ? v : o1;
        o2 = (ix == c4 + 2) ? v : o2;
        o3 = (ix == c4 + 3) ? v : o3;
      }
      F4 o = { o0, o1, o2, o3 };
      *(F4*)(orow + c4) = o;
    }
  }
}

// ---------------------------------------------------------------- launcher
extern "C" void kernel_launch(void* const* d_in, const int* in_sizes, int n_in,
                              void* d_out, int out_size, void* d_ws, size_t ws_size,
                              hipStream_t stream) {
  (void)in_sizes; (void)n_in; (void)out_size; (void)ws_size;

  const float*     x    = (const float*)d_in[0];
  const long long* idx  = (const long long*)d_in[1];
  const float*     emb1 = (const float*)d_in[2];
  const float*     emb2 = (const float*)d_in[3];
  const float*     l1w  = (const float*)d_in[4];
  const float*     l1b  = (const float*)d_in[5];
  const float*     l2w  = (const float*)d_in[6];
  const float*     l2b  = (const float*)d_in[7];
  float* out = (float*)d_out;

  // workspace layout (~51.4 MB total)
  const size_t XN = (size_t)BSZ * NN * FT;            // 8,388,608 elems
  unsigned short* xhi = (unsigned short*)d_ws;        // 16 MB
  unsigned short* xlo = xhi + XN;                     // 16 MB
  float* n1    = (float*)(xlo + XN);                  // 0.5 MB
  float* n2    = n1 + (size_t)NN * DIMN;              // 0.5 MB
  float* adjcs = n2 + (size_t)NN * DIMN;              // 16 MB

  split_bf16_kernel<<<(int)(XN / 256), 256, 0, stream>>>(x, xhi, xlo);
  node_mlp_kernel<<<NN, DIMN, 0, stream>>>(emb1, emb2, idx, l1w, l1b, l2w, l2b, n1, n2);
  adj_cs_kernel<<<dim3(NN / 16, NN / 16), dim3(16, 16), 0, stream>>>(n1, n2, adjcs);
  fused_adj_kernel<<<dim3(NN / 16, BSZ), 256, 0, stream>>>(xhi, xlo, adjcs, out);
}
